// pre_GCNModel_edge_reward_32504312496825
// MI455X (gfx1250) — compile-verified, matrix operand order checked by reading, not run on hardware
//
#include <hip/hip_runtime.h>
#include <hip/hip_bf16.h>
#include <math.h>

typedef __attribute__((ext_vector_type(16))) _Float16 v16h;
typedef __attribute__((ext_vector_type(8)))  _Float16 v8h;
typedef __attribute__((ext_vector_type(8)))  float    v8f;

static constexpr int cN  = 50000;
static constexpr int cE  = 400000;
static constexpr int cIN = 32;
static constexpr int cH  = 32;
static constexpr int cED = 16;
static constexpr int cB  = 64;
static constexpr int cOH = 256;
static constexpr int K1  = cED * cIN + cIN;  // 544: 16 chunks of 32 + bias chunk

__device__ __forceinline__ float leaky(float v) { return v > 0.f ? v : 0.05f * v; }

// Load one 16-bit B (or A) fragment half-pattern from a row-major row:
// lanes 0-15 take k in {0..7, 16..23}, lanes 16-31 take {8..15, 24..31}.
__device__ __forceinline__ v16h load_frag(const _Float16* rowChunk, int hi) {
    const _Float16* p = rowChunk + (hi ? 8 : 0);
    union { v16h v; v8h h[2]; } u;
    u.h[0] = *(const v8h*)(p);
    u.h[1] = *(const v8h*)(p + 16);
    return u.v;
}

// Gather a lane's 16-element A-row slice (pattern above) as f16.
__device__ __forceinline__ v16h load_row_f16(const float* row, int off) {
    alignas(16) float t[16];
    *(float4*)(t + 0)  = *(const float4*)(row + off);
    *(float4*)(t + 4)  = *(const float4*)(row + off + 4);
    *(float4*)(t + 8)  = *(const float4*)(row + off + 16);
    *(float4*)(t + 12) = *(const float4*)(row + off + 20);
    v16h r;
    #pragma unroll
    for (int j = 0; j < 16; ++j) r[j] = (_Float16)t[j];
    return r;
}

// ---------------- per-edge message GEMM: agg[dst] += (ea ⊗ x[src]) @ W ----------------
// One wave = 32 edges (two 16-row M tiles) x 32 outputs x K=544.
__global__ __launch_bounds__(256)
void msg_kernel(const float* __restrict__ x, const float* __restrict__ ea,
                const int* __restrict__ src, const int* __restrict__ dst,
                const _Float16* __restrict__ WtG, float* __restrict__ agg) {
    __shared__ _Float16 sW[32 * K1];           // 34 KB, [o row][k] row-major
    const int tid = threadIdx.x;
    for (int i = tid; i < (32 * K1 * 2) / 16; i += 256)
        ((uint4*)sW)[i] = ((const uint4*)WtG)[i];
    __syncthreads();

    const int lane = tid & 31, wid = tid >> 5;
    const int hi = lane >> 4, col = lane & 15;
    const int tile = blockIdx.x * 8 + wid;
    if (tile >= cE / 32) return;               // wave-uniform guard; EXEC all-ones inside
    const int e0 = tile * 32;
    const int off = hi ? 8 : 0;

    v16h xh[2];                                 // x[src] row slices, f16
    v16h eah[2];                                // edge_attr rows, f16
    #pragma unroll
    for (int t = 0; t < 2; ++t) {
        const int erow = e0 + t * 16 + col;
        xh[t] = load_row_f16(x + (size_t)src[erow] * 32, off);
        const float* er = ea + (size_t)erow * 16;
        alignas(16) float ev[16];
        *(float4*)(ev + 0)  = *(const float4*)(er + 0);
        *(float4*)(ev + 4)  = *(const float4*)(er + 4);
        *(float4*)(ev + 8)  = *(const float4*)(er + 8);
        *(float4*)(ev + 12) = *(const float4*)(er + 12);
        #pragma unroll
        for (int j = 0; j < 16; ++j) eah[t][j] = (_Float16)ev[j];
    }

    v8f c00 = {}, c01 = {}, c10 = {}, c11 = {};
    #pragma unroll
    for (int d = 0; d < 17; ++d) {              // 16 edge-attr chunks + 1 bias chunk
        v16h b0 = load_frag(sW + (size_t)col * K1 + d * 32, hi);
        v16h b1 = load_frag(sW + (size_t)(16 + col) * K1 + d * 32, hi);
        v16h a0, a1;
        if (d < 16) {
            const _Float16 s0 = eah[0][d];
            const _Float16 s1 = eah[1][d];
            a0 = xh[0] * s0;                    // v_pk_mul_f16 broadcast
            a1 = xh[1] * s1;
        } else {
            a0 = xh[0];                         // bias chunk: coefficient 1
            a1 = xh[1];
        }
        c00 = __builtin_amdgcn_wmma_f32_16x16x32_f16(false, a0, false, b0, (short)0, c00, false, false);
        c01 = __builtin_amdgcn_wmma_f32_16x16x32_f16(false, a0, false, b1, (short)0, c01, false, false);
        c10 = __builtin_amdgcn_wmma_f32_16x16x32_f16(false, a1, false, b0, (short)0, c10, false, false);
        c11 = __builtin_amdgcn_wmma_f32_16x16x32_f16(false, a1, false, b1, (short)0, c11, false, false);
    }

    #pragma unroll
    for (int r = 0; r < 8; ++r) {               // C: VGPR r -> row r + 8*hi, column col
        const int ea0 = e0 + r + (hi ? 8 : 0);
        float* ab0 = agg + (size_t)dst[ea0] * 32;
        atomicAdd(ab0 + col,      c00[r]);
        atomicAdd(ab0 + 16 + col, c01[r]);
        const int ea1 = e0 + 16 + r + (hi ? 8 : 0);
        float* ab1 = agg + (size_t)dst[ea1] * 32;
        atomicAdd(ab1 + col,      c10[r]);
        atomicAdd(ab1 + 16 + col, c11[r]);
    }
}

// ---------------- h = leaky(agg + x @ rootW + b) ----------------
__global__ __launch_bounds__(256)
void node_kernel(const float* __restrict__ xin, const float* __restrict__ agg,
                 const _Float16* __restrict__ RtG, const float* __restrict__ bias,
                 float* __restrict__ hout) {
    __shared__ _Float16 sR[32 * 32];
    const int tid = threadIdx.x;
    for (int i = tid; i < (32 * 32 * 2) / 16; i += 256)
        ((uint4*)sR)[i] = ((const uint4*)RtG)[i];
    __syncthreads();

    const int lane = tid & 31, wid = tid >> 5;
    const int hi = lane >> 4, col = lane & 15;
    const int tile = blockIdx.x * 8 + wid;
    if (tile >= cN / 16) return;
    const int n0 = tile * 16;
    const int off = hi ? 8 : 0;

    v16h a = load_row_f16(xin + (size_t)(n0 + col) * 32, off);

    v8f c0, c1;
    #pragma unroll
    for (int r = 0; r < 8; ++r) {               // seed C with scattered aggregate
        const int node = n0 + r + (hi ? 8 : 0);
        c0[r] = agg[(size_t)node * 32 + col];
        c1[r] = agg[(size_t)node * 32 + 16 + col];
    }
    v16h b0 = load_frag(sR + (size_t)col * 32, hi);
    v16h b1 = load_frag(sR + (size_t)(16 + col) * 32, hi);
    c0 = __builtin_amdgcn_wmma_f32_16x16x32_f16(false, a, false, b0, (short)0, c0, false, false);
    c1 = __builtin_amdgcn_wmma_f32_16x16x32_f16(false, a, false, b1, (short)0, c1, false, false);

    const float bb0 = bias[col], bb1 = bias[16 + col];
    #pragma unroll
    for (int r = 0; r < 8; ++r) {
        const int node = n0 + r + (hi ? 8 : 0);
        hout[(size_t)node * 32 + col]      = leaky(c0[r] + bb0);
        hout[(size_t)node * 32 + 16 + col] = leaky(c1[r] + bb1);
    }
}

// ---------------- gated readout: pooled[batch] += sig(-(hc@sW+b)) * tanh(hc@tW+b) ----------------
__global__ __launch_bounds__(256)
void gate_kernel(const float* __restrict__ h2, const float* __restrict__ x0,
                 const _Float16* __restrict__ SwG, const _Float16* __restrict__ TwG,
                 const float* __restrict__ sb, const float* __restrict__ tb,
                 const int* __restrict__ batch, float* __restrict__ pooled) {
    __shared__ _Float16 sS[32 * 64];
    __shared__ _Float16 sT[32 * 64];
    const int tid = threadIdx.x;
    for (int i = tid; i < 256; i += 256) {
        ((uint4*)sS)[i] = ((const uint4*)SwG)[i];
        ((uint4*)sT)[i] = ((const uint4*)TwG)[i];
    }
    __syncthreads();

    const int lane = tid & 31, wid = tid >> 5;
    const int hi = lane >> 4, col = lane & 15;
    const int tile = blockIdx.x * 8 + wid;
    if (tile >= cN / 16) return;
    const int n0 = tile * 16;
    const int off = hi ? 8 : 0;

    v16h ah = load_row_f16(h2 + (size_t)(n0 + col) * 32, off);
    v16h ax = load_row_f16(x0 + (size_t)(n0 + col) * 32, off);

    v8f cs0 = {}, cs1 = {}, ct0 = {}, ct1 = {};
    #pragma unroll
    for (int c = 0; c < 2; ++c) {               // K-chunk 0 = h2, chunk 1 = x0 (concat)
        v16h a = (c == 0) ? ah : ax;
        v16h bs0 = load_frag(sS + (size_t)col * 64 + c * 32, hi);
        v16h bs1 = load_frag(sS + (size_t)(16 + col) * 64 + c * 32, hi);
        v16h bt0 = load_frag(sT + (size_t)col * 64 + c * 32, hi);
        v16h bt1 = load_frag(sT + (size_t)(16 + col) * 64 + c * 32, hi);
        cs0 = __builtin_amdgcn_wmma_f32_16x16x32_f16(false, a, false, bs0, (short)0, cs0, false, false);
        cs1 = __builtin_amdgcn_wmma_f32_16x16x32_f16(false, a, false, bs1, (short)0, cs1, false, false);
        ct0 = __builtin_amdgcn_wmma_f32_16x16x32_f16(false, a, false, bt0, (short)0, ct0, false, false);
        ct1 = __builtin_amdgcn_wmma_f32_16x16x32_f16(false, a, false, bt1, (short)0, ct1, false, false);
    }

    const float sb0 = sb[col], sb1 = sb[16 + col];
    const float tb0 = tb[col], tb1 = tb[16 + col];
    #pragma unroll
    for (int r = 0; r < 8; ++r) {
        const int node = n0 + r + (hi ? 8 : 0);
        float* pb = pooled + (size_t)batch[node] * 32;
        float s0 = fminf(fmaxf(cs0[r] + sb0, -30.f), 30.f);
        float s1 = fminf(fmaxf(cs1[r] + sb1, -30.f), 30.f);
        float v0 = (1.f / (1.f + __expf(s0))) * tanhf(ct0[r] + tb0);
        float v1 = (1.f / (1.f + __expf(s1))) * tanhf(ct1[r] + tb1);
        atomicAdd(pb + col,      v0);
        atomicAdd(pb + 16 + col, v1);
    }
}

// ---------------- tiny final MLP: out = sigmoid((leaky(pooled@f1+b)@f2 + b)/4) ----------------
__global__ void final_kernel(const float* __restrict__ pooled,
                             const float* __restrict__ f1W, const float* __restrict__ f1b,
                             const float* __restrict__ f2W, const float* __restrict__ f2b,
                             float* __restrict__ out) {
    const int b = threadIdx.x;
    if (b >= cB) return;
    float pr[32];
    #pragma unroll
    for (int i = 0; i < 32; ++i) pr[i] = pooled[b * 32 + i];
    float acc = f2b[0];
    for (int j = 0; j < cOH; ++j) {
        float o = f1b[j];
        #pragma unroll
        for (int i = 0; i < 32; ++i) o += pr[i] * f1W[i * cOH + j];
        acc += leaky(o) * f2W[j];
    }
    out[b] = 1.f / (1.f + __expf(-acc * 0.25f));
}

// ---------------- prep kernels ----------------
__global__ void zero_kernel(float* p, int n) {
    int i = blockIdx.x * blockDim.x + threadIdx.x;
    if (i < n) p[i] = 0.f;
}
// nn_W [16,32,32] (+bias chunk) -> f16 Wt [o=32][k=544]
__global__ void wtconv_kernel(const float* __restrict__ W, const float* __restrict__ b,
                              _Float16* __restrict__ dst) {
    int idx = blockIdx.x * blockDim.x + threadIdx.x;
    if (idx >= 32 * K1) return;
    int o = idx / K1, k = idx % K1;
    float v;
    if (k < 512) { int d = k >> 5, i = k & 31; v = W[d * 1024 + i * 32 + o]; }
    else         { int i = k - 512;            v = b[i * 32 + o]; }
    dst[idx] = (_Float16)v;
}
// src f32 [K,32] -> f16 dst [32][K]
__global__ void tconv_kernel(const float* __restrict__ src, _Float16* __restrict__ dst, int K) {
    int idx = blockIdx.x * blockDim.x + threadIdx.x;
    if (idx >= 32 * K) return;
    int o = idx / K, k = idx % K;
    dst[idx] = (_Float16)src[k * 32 + o];
}

extern "C" void kernel_launch(void* const* d_in, const int* in_sizes, int n_in,
                              void* d_out, int out_size, void* d_ws, size_t ws_size,
                              hipStream_t stream) {
    const float* x      = (const float*)d_in[0];
    const float* ea     = (const float*)d_in[1];
    const float* nn1_W  = (const float*)d_in[2];
    const float* nn1_b  = (const float*)d_in[3];
    const float* rt1_W  = (const float*)d_in[4];
    const float* rt1_b  = (const float*)d_in[5];
    const float* nn2_W  = (const float*)d_in[6];
    const float* nn2_b  = (const float*)d_in[7];
    const float* rt2_W  = (const float*)d_in[8];
    const float* rt2_b  = (const float*)d_in[9];
    const float* sig_W  = (const float*)d_in[10];
    const float* sig_b  = (const float*)d_in[11];
    const float* tanh_W = (const float*)d_in[12];
    const float* tanh_b = (const float*)d_in[13];
    const float* f1W    = (const float*)d_in[14];
    const float* f1b    = (const float*)d_in[15];
    const float* f2W    = (const float*)d_in[16];
    const float* f2b    = (const float*)d_in[17];
    const int*   eidx   = (const int*)d_in[18];
    const int*   batch  = (const int*)d_in[19];
    const int* srcI = eidx;
    const int* dstI = eidx + cE;

    float* fws    = (float*)d_ws;
    float* agg1   = fws;                  // N*32
    float* agg2   = fws + 1600000;        // N*32
    float* h1     = fws + 3200000;        // N*32
    float* h2     = fws + 4800000;        // N*32
    float* pooled = fws + 6400000;        // B*32
    _Float16* hb  = (_Float16*)(fws + 6402048);
    _Float16* Wt1 = hb;                   // 32*544
    _Float16* Wt2 = hb + 32 * K1;
    _Float16* Rt1 = hb + 2 * 32 * K1;     // 32*32
    _Float16* Rt2 = Rt1 + 1024;
    _Float16* Sw  = Rt2 + 1024;           // 32*64
    _Float16* Tw  = Sw + 2048;

    zero_kernel<<<(3200000 + 255) / 256, 256, 0, stream>>>(agg1, 3200000); // agg1+agg2
    zero_kernel<<<8, 256, 0, stream>>>(pooled, cB * 32);
    wtconv_kernel<<<(32 * K1 + 255) / 256, 256, 0, stream>>>(nn1_W, nn1_b, Wt1);
    wtconv_kernel<<<(32 * K1 + 255) / 256, 256, 0, stream>>>(nn2_W, nn2_b, Wt2);
    tconv_kernel<<<4, 256, 0, stream>>>(rt1_W, Rt1, 32);
    tconv_kernel<<<4, 256, 0, stream>>>(rt2_W, Rt2, 32);
    tconv_kernel<<<8, 256, 0, stream>>>(sig_W, Sw, 64);
    tconv_kernel<<<8, 256, 0, stream>>>(tanh_W, Tw, 64);

    msg_kernel<<<(cE / 32 + 7) / 8, 256, 0, stream>>>(x, ea, srcI, dstI, Wt1, agg1);
    node_kernel<<<(cN / 16 + 7) / 8, 256, 0, stream>>>(x, agg1, Rt1, rt1_b, h1);
    msg_kernel<<<(cE / 32 + 7) / 8, 256, 0, stream>>>(h1, ea, srcI, dstI, Wt2, agg2);
    node_kernel<<<(cN / 16 + 7) / 8, 256, 0, stream>>>(h1, agg2, Rt2, rt2_b, h2);
    gate_kernel<<<(cN / 16 + 7) / 8, 256, 0, stream>>>(h2, x, Sw, Tw, sig_b, tanh_b, batch, pooled);
    final_kernel<<<1, 64, 0, stream>>>(pooled, f1W, f1b, f2W, f2b, (float*)d_out);
}
